// Gate_63436666962295
// MI455X (gfx1250) — matrix operand order repaired and checked by reading, not compiled, not run on hardware
//
#include <hip/hip_runtime.h>
#include <hip/hip_bf16.h>
#include <stdint.h>

// ---------------------------------------------------------------------------
// MoE gate (group-limited routing) for gfx1250.
//   scores = x[N,4096] @ weight[64,4096]^T  -> softmax -> group top-4 -> top-8
// GEMM runs on V_WMMA_F32_16X16X32_BF16 with an error-compensated bf16 split
// (x_hi*w_hi + x_hi*w_lo + x_lo*w_hi) for ~fp32 accuracy. The x-side split is
// a truncating split packed with V_PERM_B32 (2 elems/op) -- the residual term
// absorbs the truncation, keeping ~1e-5 relative accuracy on the logits.
// x is streamed with non-temporal loads (512MB, single use, > L2); bf16
// weights (1MB) stay L2-resident.
// ---------------------------------------------------------------------------

static constexpr int kDim     = 4096;
static constexpr int kExperts = 64;
static constexpr int kGroups  = 8;
static constexpr int kTopG    = 4;
static constexpr int kTopK    = 8;

typedef __attribute__((ext_vector_type(16))) __bf16 v16bf;
typedef __attribute__((ext_vector_type(8)))  float  v8f;
typedef __attribute__((ext_vector_type(4)))  float  v4f;   // clang vector: OK for nontemporal builtin

union V16 {
  v16bf          v;
  unsigned       u[8];
  unsigned short h[16];
  uint4          q[2];
};

__device__ __forceinline__ unsigned short f2bf_rn(float f) {
  unsigned u = __builtin_bit_cast(unsigned, f);
  u += 0x7FFFu + ((u >> 16) & 1u);        // round-to-nearest-even to bf16
  return (unsigned short)(u >> 16);
}
__device__ __forceinline__ float bf_up(unsigned short h) {
  return __builtin_bit_cast(float, ((unsigned)h) << 16);
}

// --- Kernel 0: split fp32 weight into bf16 hi/lo halves (64*4096 elems) ----
__global__ __launch_bounds__(256) void gate_wsplit_kernel(
    const float* __restrict__ w, unsigned short* __restrict__ whi,
    unsigned short* __restrict__ wlo, int n) {
  int i = blockIdx.x * blockDim.x + threadIdx.x;
  if (i >= n) return;
  float f = w[i];
  unsigned short h = f2bf_rn(f);
  whi[i] = h;
  wlo[i] = f2bf_rn(f - bf_up(h));
}

// --- Kernel 1: router GEMM via bf16 WMMA -----------------------------------
// One wave = one 16-token tile x all 64 experts (4 accumulator tiles).
// 8 waves per block -> 128 tokens per block.
__global__ __launch_bounds__(256) void gate_gemm_kernel(
    const float* __restrict__ x,
    const unsigned short* __restrict__ whi,
    const unsigned short* __restrict__ wlo,
    float* __restrict__ scores) {
  const int  lane  = threadIdx.x & 31;
  const int  wave  = threadIdx.x >> 5;
  const int  tileT = blockIdx.x * 8 + wave;
  const int  col   = lane & 15;       // M for A fragment, N for B/C fragments
  const bool hih   = lane >= 16;

  const float* xrow = x + (size_t)(tileT * 16 + col) * kDim;

  v8f acc[4];
#pragma unroll
  for (int t = 0; t < 4; ++t)
#pragma unroll
    for (int i = 0; i < 8; ++i) acc[t][i] = 0.0f;

  const int aoff = hih ? 8 : 0;    // A: lanes16-31 hold K chunks +8
  const int boff = hih ? 8 : 0;    // B: lanes16-31 hold K chunks +8

  for (int kb = 0; kb < kDim; kb += 32) {
    // ---- A fragment: 16 fp32 of this token row, split to bf16 hi/lo ----
    const float* ap = xrow + kb + aoff;
    v4f f0 = __builtin_nontemporal_load((const v4f*)(ap));
    v4f f1 = __builtin_nontemporal_load((const v4f*)(ap + 4));
    v4f f2 = __builtin_nontemporal_load((const v4f*)(ap + 16));
    v4f f3 = __builtin_nontemporal_load((const v4f*)(ap + 20));
    float fa[16] = {f0[0], f0[1], f0[2], f0[3], f1[0], f1[1], f1[2], f1[3],
                    f2[0], f2[1], f2[2], f2[3], f3[0], f3[1], f3[2], f3[3]};
    V16 ah, al;
#pragma unroll
    for (int i = 0; i < 8; ++i) {
      const float a = fa[2 * i], b = fa[2 * i + 1];
      const unsigned ua = __builtin_bit_cast(unsigned, a);
      const unsigned ub = __builtin_bit_cast(unsigned, b);
      // hi = truncated-to-bf16 pair, packed in one V_PERM_B32
      ah.u[i] = __builtin_amdgcn_perm(ub, ua, 0x07060302u);
      // residual (exact): d = a - float(hi)
      const float da = a - __builtin_bit_cast(float, ua & 0xffff0000u);
      const float db = b - __builtin_bit_cast(float, ub & 0xffff0000u);
      // lo = truncated-to-bf16 residual pair
      al.u[i] = __builtin_amdgcn_perm(__builtin_bit_cast(unsigned, db),
                                      __builtin_bit_cast(unsigned, da),
                                      0x07060302u);
    }

    // ---- 4 expert tiles of 16; B rows are contiguous bf16 in memory ----
#pragma unroll
    for (int te = 0; te < 4; ++te) {
      const size_t wo = (size_t)(te * 16 + col) * kDim + kb + boff;
      V16 bh, bl;
      bh.q[0] = *(const uint4*)(whi + wo);
      bh.q[1] = *(const uint4*)(whi + wo + 16);
      bl.q[0] = *(const uint4*)(wlo + wo);
      bl.q[1] = *(const uint4*)(wlo + wo + 16);
      acc[te] = __builtin_amdgcn_wmma_f32_16x16x32_bf16(
          false, ah.v, false, bh.v, (short)0, acc[te], false, false);
      acc[te] = __builtin_amdgcn_wmma_f32_16x16x32_bf16(
          false, ah.v, false, bl.v, (short)0, acc[te], false, false);
      acc[te] = __builtin_amdgcn_wmma_f32_16x16x32_bf16(
          false, al.v, false, bh.v, (short)0, acc[te], false, false);
    }
  }

  // ---- store C: VGPR r holds (M = r or r+8, N = lane&15) ----
  const int rbase = tileT * 16 + (hih ? 8 : 0);
#pragma unroll
  for (int te = 0; te < 4; ++te)
#pragma unroll
    for (int r = 0; r < 8; ++r)
      scores[(size_t)(rbase + r) * kExperts + te * 16 + col] = acc[te][r];
}

// --- Kernel 2: softmax + group-limited top-k (one thread per token) --------
__global__ __launch_bounds__(256) void gate_topk_kernel(
    const float* __restrict__ scores, float* __restrict__ w_out,
    int* __restrict__ idx_out, int n) {
  int t = blockIdx.x * blockDim.x + threadIdx.x;
  if (t >= n) return;

  float s[kExperts];
  const float4* p = (const float4*)(scores + (size_t)t * kExperts);
#pragma unroll
  for (int i = 0; i < kExperts / 4; ++i) {
    float4 v = p[i];
    s[4 * i + 0] = v.x; s[4 * i + 1] = v.y;
    s[4 * i + 2] = v.z; s[4 * i + 3] = v.w;
  }

  // softmax over all 64 experts
  float m = s[0];
#pragma unroll
  for (int i = 1; i < kExperts; ++i) m = fmaxf(m, s[i]);
  float sum = 0.0f;
#pragma unroll
  for (int i = 0; i < kExperts; ++i) { s[i] = __expf(s[i] - m); sum += s[i]; }
  float inv = 1.0f / sum;
#pragma unroll
  for (int i = 0; i < kExperts; ++i) s[i] *= inv;

  // group scores = max within each group of 8
  float gs[kGroups];
#pragma unroll
  for (int g = 0; g < kGroups; ++g) {
    float gm = s[8 * g];
#pragma unroll
    for (int j = 1; j < 8; ++j) gm = fmaxf(gm, s[8 * g + j]);
    gs[g] = gm;
  }

  // top-4 groups (strict '>' => lowest index on ties, matches lax.top_k)
  unsigned gmask = 0;
#pragma unroll
  for (int k = 0; k < kTopG; ++k) {
    float best = -1.0f; int bg = 0;
#pragma unroll
    for (int g = 0; g < kGroups; ++g) {
      bool taken = (gmask >> g) & 1u;
      if (!taken && gs[g] > best) { best = gs[g]; bg = g; }
    }
    gmask |= 1u << bg;
  }

  // mask scores by selected groups (softmax probs > 0, so -1 excludes)
  float ms[kExperts];
#pragma unroll
  for (int e = 0; e < kExperts; ++e)
    ms[e] = ((gmask >> (e >> 3)) & 1u) ? s[e] : -1.0f;

  // top-8 experts among masked
  unsigned pl = 0, ph = 0;
#pragma unroll
  for (int k = 0; k < kTopK; ++k) {
    float best = -2.0f; int bi = 0;
#pragma unroll
    for (int e = 0; e < kExperts; ++e) {
      bool taken = (e < 32) ? ((pl >> e) & 1u) : ((ph >> (e - 32)) & 1u);
      if (!taken && ms[e] > best) { best = ms[e]; bi = e; }
    }
    if (bi < 32) pl |= 1u << bi; else ph |= 1u << (bi - 32);
    w_out[(size_t)t * kTopK + k]   = best;   // ROUTE_SCALE == 1.0
    idx_out[(size_t)t * kTopK + k] = bi;
  }
}

// ---------------------------------------------------------------------------
extern "C" void kernel_launch(void* const* d_in, const int* in_sizes, int n_in,
                              void* d_out, int out_size, void* d_ws,
                              size_t ws_size, hipStream_t stream) {
  const float* x = (const float*)d_in[0];
  const float* w = (const float*)d_in[1];
  const int n_tokens = in_sizes[0] / kDim;          // 32768
  const int wn = kExperts * kDim;                   // 262144

  // workspace layout: [w_hi bf16 | w_lo bf16 | scores f32]
  unsigned short* whi = (unsigned short*)d_ws;
  unsigned short* wlo = whi + wn;
  float* scores = (float*)((char*)d_ws + (size_t)wn * 2 * sizeof(unsigned short));

  gate_wsplit_kernel<<<(wn + 255) / 256, 256, 0, stream>>>(w, whi, wlo, wn);
  gate_gemm_kernel<<<n_tokens / 128, 256, 0, stream>>>(x, whi, wlo, scores);

  float* wout = (float*)d_out;
  int*   iout = (int*)d_out + (size_t)n_tokens * kTopK;
  gate_topk_kernel<<<(n_tokens + 255) / 256, 256, 0, stream>>>(scores, wout,
                                                               iout, n_tokens);
}
